// GraphAttentionNetwork_26757646254096
// MI455X (gfx1250) — matrix operand order repaired and checked by reading, not compiled, not run on hardware
//
#include <hip/hip_runtime.h>

#define N_NODES 8192
#define IN_DIM  512
#define HID     256
#define OUT_DIM 64
#define NEG_INF (-9.0e15f)

typedef __attribute__((ext_vector_type(2)))  __bf16 v2bf;
typedef __attribute__((ext_vector_type(8)))  __bf16 v8bf;
typedef __attribute__((ext_vector_type(16))) __bf16 v16bf;
typedef __attribute__((ext_vector_type(4)))  float  v4f;
typedef __attribute__((ext_vector_type(8)))  float  v8f;

// Combine two 8-element bf16 chunks into a 16-element WMMA operand register set.
static __device__ inline v16bf make16(const __bf16* lo, const __bf16* hi) {
  v8bf l = *(const v8bf*)lo;
  v8bf h = *(const v8bf*)hi;
  v16bf r;
#pragma unroll
  for (int i = 0; i < 8; ++i) { r[i] = l[i]; r[i + 8] = h[i]; }
  return r;
}

static __device__ inline v16bf join16(v8bf l, v8bf h) {
  v16bf r;
#pragma unroll
  for (int i = 0; i < 8; ++i) { r[i] = l[i]; r[i + 8] = h[i]; }
  return r;
}

// ---------------------------------------------------------------------------
// Kernel 0a: W1 (512x256 f32) -> W1T (256x512 bf16): contiguous B-operand loads.
// ---------------------------------------------------------------------------
__global__ __launch_bounds__(256) void w1t_kernel(const float* __restrict__ W1,
                                                  __bf16* __restrict__ W1T) {
  int idx = blockIdx.x * 256 + threadIdx.x;   // 0..131071
  int k = idx >> 8;                            // 0..511
  int n = idx & 255;                           // 0..255
  W1T[n * IN_DIM + k] = (__bf16)W1[idx];
}

// ---------------------------------------------------------------------------
// Kernel 0b: Omega (256x64 f32) -> OmegaT (64x256 bf16) for the output head.
// ---------------------------------------------------------------------------
__global__ __launch_bounds__(256) void omegat_kernel(
    const float* __restrict__ Omega, __bf16* __restrict__ OmegaT) {
  int idx = blockIdx.x * 256 + threadIdx.x;   // 0..16383
  int n = idx >> 6;                            // 0..255
  int o = idx & 63;                            // 0..63
  OmegaT[o * HID + n] = (__bf16)Omega[idx];
}

// ---------------------------------------------------------------------------
// Kernel 1: H1 = X @ W1 + b1 (bf16 WMMA, fp32 accum).
//   - writes H1 transposed as bf16 (H1T[256][8192]) for kernel 2 B-loads
//   - computes s1 = H1 @ a[:256], s2 = H1 @ a[256:] deterministically
// Double-buffered X staging: ONE barrier per K-chunk.
// ---------------------------------------------------------------------------
__global__ __launch_bounds__(256) void h1_kernel(
    const float* __restrict__ X, const float* __restrict__ avec,
    const float* __restrict__ b1, const __bf16* __restrict__ W1T,
    __bf16* __restrict__ H1T, float* __restrict__ s1, float* __restrict__ s2) {
  __shared__ __attribute__((aligned(16))) __bf16 sX[2][16 * 32];
  __shared__ float sPart1[16][8];
  __shared__ float sPart2[16][8];

  const int tid  = threadIdx.x;
  const int wave = tid >> 5;
  const int lane = tid & 31;
  const int m0   = blockIdx.x * 16;
  const int lcol = lane & 15;
  const int half = (lane < 16) ? 0 : 8;
  const int aklo = (lane < 16) ? 0 : 16;
  const int bklo = (lane < 16) ? 0 : 16;
  const int n0   = wave * 32;
  const int xr = tid >> 4;          // 0..15
  const int xc = (tid & 15) * 2;    // 0,2,..,30

  v8f acc0 = {};
  v8f acc1 = {};
  int buf = 0;

  for (int kc = 0; kc < IN_DIM; kc += 32) {
    {  // packed bf16 staging: float2 load -> cvt_pk -> ds_store_b32
      float2 xv = *(const float2*)&X[(size_t)(m0 + xr) * IN_DIM + kc + xc];
      v2bf p; p[0] = (__bf16)xv.x; p[1] = (__bf16)xv.y;
      *(v2bf*)&sX[buf][xr * 32 + xc] = p;
    }
    // B-operand loads issued before the barrier (independent of LDS).
    const __bf16* bp0 = &W1T[(size_t)(n0 + lcol) * IN_DIM + kc + bklo];
    v8bf b0l = *(const v8bf*)bp0;
    v8bf b0h = *(const v8bf*)(bp0 + 8);
    const __bf16* bp1 = &W1T[(size_t)(n0 + 16 + lcol) * IN_DIM + kc + bklo];
    v8bf b1l = *(const v8bf*)bp1;
    v8bf b1h = *(const v8bf*)(bp1 + 8);

    __syncthreads();

    v16bf Am = make16(&sX[buf][lcol * 32 + aklo],
                      &sX[buf][lcol * 32 + aklo + 8]);
    acc0 = __builtin_amdgcn_wmma_f32_16x16x32_bf16(
        false, Am, false, join16(b0l, b0h), (short)0, acc0, false, false);
    acc1 = __builtin_amdgcn_wmma_f32_16x16x32_bf16(
        false, Am, false, join16(b1l, b1h), (short)0, acc1, false, false);
    buf ^= 1;
  }

  const float bias0 = b1[n0 + lcol];
  const float bias1 = b1[n0 + 16 + lcol];
  const float a1v0  = avec[n0 + lcol];
  const float a2v0  = avec[HID + n0 + lcol];
  const float a1v1  = avec[n0 + 16 + lcol];
  const float a2v1  = avec[HID + n0 + 16 + lcol];

  v8bf pack0, pack1;
#pragma unroll
  for (int r = 0; r < 8; ++r) {
    float h0 = acc0[r] + bias0;
    float h1 = acc1[r] + bias1;
    pack0[r] = (__bf16)h0;
    pack1[r] = (__bf16)h1;
    float v1 = h0 * a1v0 + h1 * a1v1;
    float v2 = h0 * a2v0 + h1 * a2v1;
#pragma unroll
    for (int mask = 8; mask >= 1; mask >>= 1) {  // reduce within 16-lane half
      v1 += __shfl_xor(v1, mask, 32);
      v2 += __shfl_xor(v2, mask, 32);
    }
    if (lcol == 0) {
      sPart1[half + r][wave] = v1;
      sPart2[half + r][wave] = v2;
    }
  }

  // H1T[col][m0+half .. +8) is this lane's C column -> one 16B store each.
  *(v8bf*)&H1T[(size_t)(n0 + lcol) * N_NODES + m0 + half]      = pack0;
  *(v8bf*)&H1T[(size_t)(n0 + 16 + lcol) * N_NODES + m0 + half] = pack1;

  __syncthreads();
  if (tid < 16) {  // fixed-order reduction: deterministic
    float t1 = 0.f, t2 = 0.f;
#pragma unroll
    for (int w = 0; w < 8; ++w) { t1 += sPart1[tid][w]; t2 += sPart2[tid][w]; }
    s1[m0 + tid] = t1;
    s2[m0 + tid] = t2;
  }
}

// ---------------------------------------------------------------------------
// Kernel 2: fused masked-softmax attention + aggregation + WMMA output head.
// Per block: 16 rows; stream 8192 cols in 64-wide chunks with online softmax.
// Double-buffered P tile / scale factors: ONE barrier per chunk.
// A-mask is streamed non-temporally (read exactly once; keep H1T in L2).
// ---------------------------------------------------------------------------
__global__ __launch_bounds__(256) void attn_kernel(
    const int* __restrict__ A, const float* __restrict__ s1g,
    const float* __restrict__ s2g, const __bf16* __restrict__ H1T,
    const __bf16* __restrict__ OmegaT, const float* __restrict__ beta,
    float* __restrict__ out) {
  __shared__ __attribute__((aligned(16))) __bf16 sP[2][16 * 64];
  __shared__ __attribute__((aligned(16))) __bf16 sH2b[16][HID];
  __shared__ __attribute__((aligned(16))) float sS1[16];
  __shared__ __attribute__((aligned(16))) float sM[16];
  __shared__ __attribute__((aligned(16))) float sL[16];
  __shared__ __attribute__((aligned(16))) float sScale[2][16];

  const int tid  = threadIdx.x;
  const int wave = tid >> 5;
  const int lane = tid & 31;
  const int i0   = blockIdx.x * 16;
  const int lcol = lane & 15;
  const int half = (lane < 16) ? 0 : 8;
  const int aklo = (lane < 16) ? 0 : 16;
  const int bklo = (lane < 16) ? 0 : 16;
  const int jj   = lane * 2;  // this lane's column pair inside a 64-wide chunk

  if (tid < 16) { sS1[tid] = s1g[i0 + tid]; sM[tid] = NEG_INF; sL[tid] = 0.f; }

  const long* ArowNT0 = (const long*)(A + (size_t)(i0 + wave) * N_NODES);
  const long* ArowNT1 = (const long*)(A + (size_t)(i0 + wave + 8) * N_NODES);

  v8f acc0 = {};
  v8f acc1 = {};
  int buf = 0;

  __syncthreads();  // sS1/sM/sL visible to all waves before first chunk

  for (int j0 = 0; j0 < N_NODES; j0 += 64) {
    // ---- B-operand loads for this chunk: independent of LDS, issue early ---
    v8bf bv[8];
#pragma unroll
    for (int ks = 0; ks < 2; ++ks) {
      const __bf16* bp0 =
          &H1T[(size_t)(wave * 32 + lcol) * N_NODES + j0 + ks * 32 + bklo];
      bv[ks * 4 + 0] = *(const v8bf*)bp0;
      bv[ks * 4 + 1] = *(const v8bf*)(bp0 + 8);
      const __bf16* bp1 =
          &H1T[(size_t)(wave * 32 + 16 + lcol) * N_NODES + j0 + ks * 32 + bklo];
      bv[ks * 4 + 2] = *(const v8bf*)bp1;
      bv[ks * 4 + 3] = *(const v8bf*)(bp1 + 8);
    }

    // ---- score phase: wave w handles rows w and w+8, 2 cols per lane ----
    float2 s2p = *(const float2*)&s2g[j0 + jj];
    long aw0 = __builtin_nontemporal_load(&ArowNT0[(j0 + jj) >> 1]);
    long aw1 = __builtin_nontemporal_load(&ArowNT1[(j0 + jj) >> 1]);
    int a0x = (int)aw0, a0y = (int)(aw0 >> 32);
    int a1x = (int)aw1, a1y = (int)(aw1 >> 32);
    float s1a = sS1[wave], s1b = sS1[wave + 8];

    float e00 = s1a + s2p.x;  e00 = (e00 > 0.f) ? e00 : 0.2f * e00;
    float e01 = s1a + s2p.y;  e01 = (e01 > 0.f) ? e01 : 0.2f * e01;
    float e10 = s1b + s2p.x;  e10 = (e10 > 0.f) ? e10 : 0.2f * e10;
    float e11 = s1b + s2p.y;  e11 = (e11 > 0.f) ? e11 : 0.2f * e11;
    e00 = (a0x > 0) ? e00 : NEG_INF;
    e01 = (a0y > 0) ? e01 : NEG_INF;
    e10 = (a1x > 0) ? e10 : NEG_INF;
    e11 = (a1y > 0) ? e11 : NEG_INF;

    float c0 = fmaxf(e00, e01), c1 = fmaxf(e10, e11);
#pragma unroll
    for (int mask = 16; mask >= 1; mask >>= 1) {
      c0 = fmaxf(c0, __shfl_xor(c0, mask, 32));
      c1 = fmaxf(c1, __shfl_xor(c1, mask, 32));
    }
    float mp0 = sM[wave], mp1 = sM[wave + 8];
    float mn0 = fmaxf(mp0, c0), mn1 = fmaxf(mp1, c1);
    float p00 = (a0x > 0) ? __expf(e00 - mn0) : 0.f;
    float p01 = (a0y > 0) ? __expf(e01 - mn0) : 0.f;
    float p10 = (a1x > 0) ? __expf(e10 - mn1) : 0.f;
    float p11 = (a1y > 0) ? __expf(e11 - mn1) : 0.f;
    {  // packed bf16 stores of the probability tile
      v2bf pk0; pk0[0] = (__bf16)p00; pk0[1] = (__bf16)p01;
      v2bf pk1; pk1[0] = (__bf16)p10; pk1[1] = (__bf16)p11;
      *(v2bf*)&sP[buf][wave * 64 + jj]       = pk0;
      *(v2bf*)&sP[buf][(wave + 8) * 64 + jj] = pk1;
    }
    float q0 = p00 + p01, q1 = p10 + p11;
#pragma unroll
    for (int mask = 16; mask >= 1; mask >>= 1) {
      q0 += __shfl_xor(q0, mask, 32);
      q1 += __shfl_xor(q1, mask, 32);
    }
    if (lane == 0) {
      float sc0 = __expf(mp0 - mn0);
      float sc1 = __expf(mp1 - mn1);
      sM[wave] = mn0;              sM[wave + 8] = mn1;
      sScale[buf][wave] = sc0;     sScale[buf][wave + 8] = sc1;
      sL[wave] = sL[wave] * sc0 + q0;
      sL[wave + 8] = sL[wave + 8] * sc1 + q1;
    }

    __syncthreads();  // the ONLY barrier per chunk

    // ---- online-softmax rescale (two ds_load_b128) ----
    v4f scl = *(const v4f*)&sScale[buf][half];
    v4f sch = *(const v4f*)&sScale[buf][half + 4];
#pragma unroll
    for (int r = 0; r < 4; ++r) { acc0[r] *= scl[r]; acc1[r] *= scl[r]; }
#pragma unroll
    for (int r = 0; r < 4; ++r) { acc0[4 + r] *= sch[r]; acc1[4 + r] *= sch[r]; }

    // ---- P (16x64) @ H1[j0:j0+64, wave cols]: 2 K-steps x 2 col tiles ----
#pragma unroll
    for (int ks = 0; ks < 2; ++ks) {
      v16bf Am = make16(&sP[buf][lcol * 64 + ks * 32 + aklo],
                        &sP[buf][lcol * 64 + ks * 32 + aklo + 8]);
      acc0 = __builtin_amdgcn_wmma_f32_16x16x32_bf16(
          false, Am, false, join16(bv[ks * 4 + 0], bv[ks * 4 + 1]), (short)0,
          acc0, false, false);
      acc1 = __builtin_amdgcn_wmma_f32_16x16x32_bf16(
          false, Am, false, join16(bv[ks * 4 + 2], bv[ks * 4 + 3]), (short)0,
          acc1, false, false);
    }
    buf ^= 1;
  }

  __syncthreads();
  // ---- normalize by softmax denominator, stage H2 as bf16 in LDS ----
  {
    v4f ll = *(const v4f*)&sL[half];
    v4f lh = *(const v4f*)&sL[half + 4];
#pragma unroll
    for (int r = 0; r < 8; ++r) {
      float lv = (r < 4) ? ll[r] : lh[r - 4];
      float linv = 1.0f / fmaxf(lv, 1e-30f);
      sH2b[half + r][wave * 32 + lcol]      = (__bf16)(acc0[r] * linv);
      sH2b[half + r][wave * 32 + 16 + lcol] = (__bf16)(acc1[r] * linv);
    }
  }
  __syncthreads();

  // ---- WMMA output head: out = sigmoid(H2 @ Omega + beta), 16x64/block ----
  if (wave < 4) {
    const int ocol = wave * 16 + lcol;   // output column for this lane
    v8f oacc = {};
#pragma unroll
    for (int kc = 0; kc < HID; kc += 32) {
      v16bf Am = make16(&sH2b[lcol][kc + aklo], &sH2b[lcol][kc + aklo + 8]);
      const __bf16* bp = &OmegaT[(size_t)ocol * HID + kc + bklo];
      v16bf Bm = make16(bp, bp + 8);
      oacc = __builtin_amdgcn_wmma_f32_16x16x32_bf16(false, Am, false, Bm,
                                                     (short)0, oacc, false,
                                                     false);
    }
    float bv = beta[ocol];
#pragma unroll
    for (int r = 0; r < 8; ++r) {
      float x = oacc[r] + bv;
      out[(size_t)(i0 + half + r) * OUT_DIM + ocol] =
          1.0f / (1.0f + __expf(-x));
    }
  }
}

// ---------------------------------------------------------------------------
extern "C" void kernel_launch(void* const* d_in, const int* in_sizes, int n_in,
                              void* d_out, int out_size, void* d_ws,
                              size_t ws_size, hipStream_t stream) {
  (void)in_sizes; (void)n_in; (void)out_size; (void)ws_size;
  const float* X     = (const float*)d_in[0];
  const int*   A     = (const int*)d_in[1];
  const float* W1    = (const float*)d_in[2];
  const float* b1    = (const float*)d_in[3];
  const float* avec  = (const float*)d_in[4];
  const float* Omega = (const float*)d_in[5];
  const float* beta  = (const float*)d_in[6];
  float* out = (float*)d_out;

  // Workspace layout (256B-aligned regions):
  //   [0, 256KB)        : W1T  bf16 (256 x 512)
  //   [256KB, +4MB)     : H1T  bf16 (256 x 8192)
  //   then              : s1[8192], s2[8192] f32
  //   then              : OmegaT bf16 (64 x 256)
  char* ws = (char*)d_ws;
  __bf16* W1T = (__bf16*)ws;
  __bf16* H1T = (__bf16*)(ws + (size_t)256 * 1024);
  float*  s1  = (float*)(ws + (size_t)256 * 1024 + (size_t)HID * N_NODES * 2);
  float*  s2  = s1 + N_NODES;
  __bf16* OmegaT = (__bf16*)(s2 + N_NODES);

  w1t_kernel<<<IN_DIM * HID / 256, 256, 0, stream>>>(W1, W1T);
  omegat_kernel<<<HID * OUT_DIM / 256, 256, 0, stream>>>(Omega, OmegaT);
  h1_kernel<<<N_NODES / 16, 256, 0, stream>>>(X, avec, b1, W1T, H1T, s1, s2);
  attn_kernel<<<N_NODES / 16, 256, 0, stream>>>(A, s1, s2, H1T, OmegaT, beta,
                                                out);
}
